// DeepSeekV3StyleMoE_71640054497668
// MI455X (gfx1250) — matrix operand order repaired and checked by reading, not compiled, not run on hardware
//
#include <hip/hip_runtime.h>
#include <cstddef>
#include <cstdint>

// ---------------- problem constants ----------------
#define T_TOK   4096      // B*S
#define H_DIM   2048
#define F_MOE_  1024
#define F_SH    2048
#define NEXP    16
#define TOPK    4
#define NGRP    4
#define EPG     4         // experts per group
#define RSCALE  2.5f
#define LDK     40        // 32 K-halves + 8 pad (row = 80B, 16B-aligned chunks)

typedef __attribute__((ext_vector_type(16))) __bf16 v16bf;
typedef __attribute__((ext_vector_type(8)))  float  v8f;

// ---------------- CDNA5 async global->LDS helpers ----------------
// GLOBAL_LOAD_ASYNC_TO_LDS_B128: each lane copies 16B from its global address to LDS at byte offset vdst.
// Tracked by ASYNCcnt (not LOADcnt/DScnt), so we wait explicitly.
__device__ __forceinline__ uint32_t lds_off(const void* p) {
  // generic LDS pointer: addr[31:0] is the LDS byte address (aperture in high bits)
  return (uint32_t)(uintptr_t)p;
}
__device__ __forceinline__ void async_cp16(uint32_t lds_byte_off, const void* gsrc) {
#if defined(__AMDGCN__)
  asm volatile("global_load_async_to_lds_b128 %0, %1, off"
               :: "v"(lds_byte_off), "v"((unsigned long long)(uintptr_t)gsrc)
               : "memory");
#endif
}
__device__ __forceinline__ void async_wait_all() {
#if defined(__AMDGCN__)
  asm volatile("s_wait_asynccnt 0" ::: "memory");
#endif
}

// ---------------- WMMA helpers ----------------
__device__ __forceinline__ v8f wmma_bf16(v16bf a, v16bf b, v8f c) {
  return __builtin_amdgcn_wmma_f32_16x16x32_bf16(false, a, false, b, (short)0, c,
                                                 false, false);
}

// A-frag (16x32 bf16): lane<16 -> M=lane, K chunks {0..7,16..23}; lane>=16 -> {8..15,24..31}
__device__ __forceinline__ v16bf lds_fragA(const __bf16* tile, int row, int lane) {
  int half = lane >> 4;
  const __bf16* p = tile + row * LDK + half * 8;
  union { v16bf v; uint4 q[2]; } u;
  u.q[0] = *(const uint4*)(p);
  u.q[1] = *(const uint4*)(p + 16);
  return u.v;
}
// B-frag (32x16 bf16): lane<16 -> N=lane, K chunks {0..7,16..23}; lane>=16 -> {8..15,24..31}
__device__ __forceinline__ v16bf lds_fragB(const __bf16* tile, int row, int lane) {
  int half = lane >> 4;
  const __bf16* p = tile + row * LDK + half * 8;
  union { v16bf v; uint4 q[2]; } u;
  u.q[0] = *(const uint4*)(p);
  u.q[1] = *(const uint4*)(p + 16);
  return u.v;
}

// ---------------- fp32 -> bf16 one-time conversion ----------------
__global__ void __launch_bounds__(256)
cvt_f32_bf16_kernel(const float* __restrict__ src, __bf16* __restrict__ dst, long n4) {
  long i = (long)blockIdx.x * 256 + threadIdx.x;
  long stride = (long)gridDim.x * 256;
  for (; i < n4; i += stride) {
    float4 x = ((const float4*)src)[i];
    union { __bf16 h[4]; uint2 q; } cv;
    cv.h[0] = (__bf16)x.x; cv.h[1] = (__bf16)x.y;
    cv.h[2] = (__bf16)x.z; cv.h[3] = (__bf16)x.w;
    ((uint2*)dst)[i] = cv.q;
  }
}

// ---------------- 1) router ----------------
__global__ void __launch_bounds__(256)
moe_router_kernel(const float* __restrict__ hidden,
                  const float* __restrict__ gate_weight,  // [16][H]
                  const float* __restrict__ bias,         // [16]
                  float* __restrict__ topk_w,             // [T][4] scaled
                  int*   __restrict__ topk_idx)           // [T][4]
{
  __shared__ float hrow[H_DIM];
  __shared__ float red[NEXP][16];
  __shared__ float logits[NEXP];
  int t   = blockIdx.x;
  int tid = threadIdx.x;
  const float* hp = hidden + (size_t)t * H_DIM;
#pragma unroll
  for (int i = 0; i < H_DIM / 256; ++i) hrow[tid + 256 * i] = hp[tid + 256 * i];
  __syncthreads();
  int e = tid >> 4, l = tid & 15;
  const float* wp = gate_weight + (size_t)e * H_DIM;
  float p = 0.f;
  for (int j = l; j < H_DIM; j += 16) p += hrow[j] * wp[j];
  red[e][l] = p;
  __syncthreads();
  if (tid < NEXP) {
    float s = 0.f;
#pragma unroll
    for (int j = 0; j < 16; ++j) s += red[tid][j];
    logits[tid] = s;
  }
  __syncthreads();
  if (tid == 0) {
    float sg[NEXP], sc[NEXP];
    for (int i = 0; i < NEXP; ++i) {
      sg[i] = 1.f / (1.f + __expf(-logits[i]));   // sigmoid scores
      sc[i] = sg[i] + bias[i];                    // scores_for_choice
    }
    float gs[NGRP];
    for (int g = 0; g < NGRP; ++g) {              // sum of top-2 per group
      float m1 = -1e30f, m2 = -1e30f;
      for (int j = 0; j < EPG; ++j) {
        float v = sc[g * EPG + j];
        if (v > m1) { m2 = m1; m1 = v; }
        else if (v > m2) m2 = v;
      }
      gs[g] = m1 + m2;
    }
    int g1 = 0;
    for (int g = 1; g < NGRP; ++g) if (gs[g] > gs[g1]) g1 = g;
    int g2 = -1;
    for (int g = 0; g < NGRP; ++g) {
      if (g == g1) continue;
      if (g2 < 0 || gs[g] > gs[g2]) g2 = g;
    }
    bool allow[NEXP];
    for (int i = 0; i < NEXP; ++i) {
      int g = i / EPG; allow[i] = (g == g1) || (g == g2);
    }
    int idx[TOPK]; float wv[TOPK];
    bool taken[NEXP] = {};
    for (int k = 0; k < TOPK; ++k) {
      int bi = -1; float bv = -1e30f;
      for (int i = 0; i < NEXP; ++i) {
        if (!allow[i] || taken[i]) continue;
        if (sc[i] > bv) { bv = sc[i]; bi = i; }
      }
      taken[bi] = true; idx[k] = bi; wv[k] = sg[bi];
    }
    float sum = 1e-20f;
    for (int k = 0; k < TOPK; ++k) sum += wv[k];
    float inv = RSCALE / sum;
    for (int k = 0; k < TOPK; ++k) {
      topk_idx[t * TOPK + k] = idx[k];
      topk_w[t * TOPK + k]   = wv[k] * inv;
    }
  }
}

// ---------------- 2) dispatch build ----------------
__global__ void moe_zero_counts(int* counts) {
  if (threadIdx.x < NEXP) counts[threadIdx.x] = 0;
}
__global__ void moe_count(const int* __restrict__ topk_idx, int* __restrict__ counts) {
  int i = blockIdx.x * 256 + threadIdx.x;
  if (i < T_TOK * TOPK) atomicAdd(&counts[topk_idx[i]], 1);
}
__global__ void moe_scan(const int* __restrict__ counts, int* __restrict__ offs,
                         int* __restrict__ cursor) {
  if (threadIdx.x == 0) {
    int o = 0;
    for (int e = 0; e < NEXP; ++e) { offs[e] = o; cursor[e] = o; o += counts[e]; }
    offs[NEXP] = o;   // == T_TOK*TOPK
  }
}
__global__ void moe_fill(const int* __restrict__ topk_idx, const float* __restrict__ topk_w,
                         int* __restrict__ cursor, int* __restrict__ slot_token,
                         float* __restrict__ slot_w, int* __restrict__ slot_of) {
  int i = blockIdx.x * 256 + threadIdx.x;
  if (i >= T_TOK * TOPK) return;
  int e   = topk_idx[i];
  int pos = atomicAdd(&cursor[e], 1);
  slot_token[pos] = i >> 2;
  slot_w[pos]     = topk_w[i];
  slot_of[i]      = pos;
}

// ---------------- 3) gate+up (+silu) GEMM, async double-buffered ----------------
__global__ void __launch_bounds__(256)
moe_gateup_kernel(const __bf16* __restrict__ hidden,   // bf16 copy of hidden
                  const __bf16* __restrict__ gw_base,  // bf16 [.,F,H]
                  const __bf16* __restrict__ uw_base,
                  const int* __restrict__ slot_token,
                  const int* __restrict__ offs,
                  __bf16* __restrict__ actOut,         // [rows][F]
                  int F, int gatherMode)
{
  int rowBase, rowEnd;
  const __bf16* gw = gw_base;
  const __bf16* uw = uw_base;
  if (gatherMode) {
    int e  = blockIdx.z;
    int s0 = offs[e], s1 = offs[e + 1];
    rowBase = s0 + 64 * blockIdx.y;
    rowEnd  = s1;
    if (rowBase >= s1) return;       // uniform exit, EXEC stays full for WMMA
    gw += (size_t)e * F * H_DIM;
    uw += (size_t)e * F * H_DIM;
  } else {
    rowBase = 64 * blockIdx.y; rowEnd = T_TOK;
  }
  int f0 = 64 * blockIdx.x;

  __shared__ __align__(16) __bf16 At[2][64 * LDK];
  __shared__ __align__(16) __bf16 Bg[2][64 * LDK];
  __shared__ __align__(16) __bf16 Bu[2][64 * LDK];

  int tid = threadIdx.x, lane = tid & 31, w = tid >> 5;
  int lrow = tid >> 2, c8 = (tid & 3) * 8;

  int arow = rowBase + lrow; if (arow > rowEnd - 1) arow = rowEnd - 1;
  int tok  = gatherMode ? slot_token[arow] : arow;
  const __bf16* aSrc = hidden + (size_t)tok * H_DIM + c8;
  const __bf16* gSrc = gw + (size_t)(f0 + lrow) * H_DIM + c8;
  const __bf16* uSrc = uw + (size_t)(f0 + lrow) * H_DIM + c8;
  uint32_t aD[2] = { lds_off(&At[0][lrow * LDK + c8]), lds_off(&At[1][lrow * LDK + c8]) };
  uint32_t gD[2] = { lds_off(&Bg[0][lrow * LDK + c8]), lds_off(&Bg[1][lrow * LDK + c8]) };
  uint32_t uD[2] = { lds_off(&Bu[0][lrow * LDK + c8]), lds_off(&Bu[1][lrow * LDK + c8]) };

  int mt = w & 3, nt = w >> 2;          // 4 m-tiles x 2 n-pairs of waves
  int aR  = 16 * mt + (lane & 15);
  int bR0 = 32 * nt + (lane & 15);
  int bR1 = bR0 + 16;

  v8f ag0 = {0.f,0.f,0.f,0.f,0.f,0.f,0.f,0.f};
  v8f ag1 = ag0, au0 = ag0, au1 = ag0;

  // prologue: fill buffer 0 via async DMA
  async_cp16(aD[0], aSrc);
  async_cp16(gD[0], gSrc);
  async_cp16(uD[0], uSrc);
  async_wait_all();
  __syncthreads();

  int p = 0;
  for (int k0 = 0; k0 < H_DIM; k0 += 32) {
    int q = p ^ 1;
    if (k0 + 32 < H_DIM) {            // fill next buffer while computing this one
      async_cp16(aD[q], aSrc + k0 + 32);
      async_cp16(gD[q], gSrc + k0 + 32);
      async_cp16(uD[q], uSrc + k0 + 32);
    }
    v16bf a  = lds_fragA(At[p], aR, lane);
    v16bf b0 = lds_fragB(Bg[p], bR0, lane);
    v16bf b1 = lds_fragB(Bg[p], bR1, lane);
    v16bf c0 = lds_fragB(Bu[p], bR0, lane);
    v16bf c1 = lds_fragB(Bu[p], bR1, lane);
    ag0 = wmma_bf16(a, b0, ag0);
    ag1 = wmma_bf16(a, b1, ag1);
    au0 = wmma_bf16(a, c0, au0);
    au1 = wmma_bf16(a, c1, au1);
    async_wait_all();                 // own async writes landed in LDS
    __syncthreads();                  // everyone's writes visible, reads done
    p = q;
  }

  int half = lane >> 4, nl = lane & 15;
#pragma unroll
  for (int j = 0; j < 2; ++j) {
    v8f g = j ? ag1 : ag0;
    v8f u = j ? au1 : au0;
#pragma unroll
    for (int r = 0; r < 8; ++r) {
      int m    = 16 * mt + 8 * half + r;   // C layout: m = vgpr + 8*(lane>=16)
      int srow = rowBase + m;
      if (srow < rowEnd) {
        float gg = (float)g[r];
        float av = gg * (1.f / (1.f + __expf(-gg))) * (float)u[r];  // silu(g)*u
        int col = f0 + 32 * nt + 16 * j + nl;
        actOut[(size_t)srow * F + col] = (__bf16)av;
      }
    }
  }
}

// ---------------- 4) routed down-proj -> weighted contrib[slot][H] ----------------
__global__ void __launch_bounds__(256)
moe_down_kernel(const __bf16* __restrict__ act,       // [slots][F_MOE]
                const __bf16* __restrict__ dw_base,   // bf16 [N][H][F_MOE]
                const float* __restrict__ slot_w,
                const int* __restrict__ offs,
                float* __restrict__ contrib)          // [slots][H]
{
  int e  = blockIdx.z;
  int s0 = offs[e], s1 = offs[e + 1];
  int rowBase = s0 + 64 * blockIdx.y;
  if (rowBase >= s1) return;
  const __bf16* dw = dw_base + (size_t)e * H_DIM * F_MOE_;
  int h0 = 64 * blockIdx.x;

  __shared__ __align__(16) __bf16 At[2][64 * LDK];
  __shared__ __align__(16) __bf16 Bt[2][64 * LDK];

  int tid = threadIdx.x, lane = tid & 31, w = tid >> 5;
  int lrow = tid >> 2, c8 = (tid & 3) * 8;
  int arow = rowBase + lrow; if (arow > s1 - 1) arow = s1 - 1;
  const __bf16* aSrc = act + (size_t)arow * F_MOE_ + c8;
  const __bf16* bSrc = dw  + (size_t)(h0 + lrow) * F_MOE_ + c8;
  uint32_t aD[2] = { lds_off(&At[0][lrow * LDK + c8]), lds_off(&At[1][lrow * LDK + c8]) };
  uint32_t bD[2] = { lds_off(&Bt[0][lrow * LDK + c8]), lds_off(&Bt[1][lrow * LDK + c8]) };

  int mt = w & 3, nt = w >> 2;
  int aR  = 16 * mt + (lane & 15);
  int bR0 = 32 * nt + (lane & 15);
  int bR1 = bR0 + 16;

  v8f a0 = {0.f,0.f,0.f,0.f,0.f,0.f,0.f,0.f};
  v8f a1 = a0;

  async_cp16(aD[0], aSrc);
  async_cp16(bD[0], bSrc);
  async_wait_all();
  __syncthreads();

  int p = 0;
  for (int k0 = 0; k0 < F_MOE_; k0 += 32) {
    int q = p ^ 1;
    if (k0 + 32 < F_MOE_) {
      async_cp16(aD[q], aSrc + k0 + 32);
      async_cp16(bD[q], bSrc + k0 + 32);
    }
    v16bf a  = lds_fragA(At[p], aR, lane);
    v16bf b0 = lds_fragB(Bt[p], bR0, lane);
    v16bf b1 = lds_fragB(Bt[p], bR1, lane);
    a0 = wmma_bf16(a, b0, a0);
    a1 = wmma_bf16(a, b1, a1);
    async_wait_all();
    __syncthreads();
    p = q;
  }
  int half = lane >> 4, nl = lane & 15;
#pragma unroll
  for (int j = 0; j < 2; ++j) {
    v8f c = j ? a1 : a0;
#pragma unroll
    for (int r = 0; r < 8; ++r) {
      int m    = 16 * mt + 8 * half + r;
      int srow = rowBase + m;
      if (srow < s1) {
        float wgt = slot_w[srow];
        int col = h0 + 32 * nt + 16 * j + nl;
        contrib[(size_t)srow * H_DIM + col] = wgt * c[r];
      }
    }
  }
}

// ---------------- 5) shared down-proj + deterministic routed sum -> out ----------------
__global__ void __launch_bounds__(256)
moe_final_kernel(const __bf16* __restrict__ actsh,    // [T][F_SH]
                 const __bf16* __restrict__ shdw,     // bf16 [H][F_SH]
                 const float* __restrict__ contrib,   // [slots][H]
                 const int* __restrict__ slot_of,     // [T][4]
                 float* __restrict__ out)             // [T][H]
{
  int rowBase = 64 * blockIdx.y;
  int h0 = 64 * blockIdx.x;

  __shared__ __align__(16) __bf16 At[2][64 * LDK];
  __shared__ __align__(16) __bf16 Bt[2][64 * LDK];

  int tid = threadIdx.x, lane = tid & 31, w = tid >> 5;
  int lrow = tid >> 2, c8 = (tid & 3) * 8;
  const __bf16* aSrc = actsh + (size_t)(rowBase + lrow) * F_SH + c8;
  const __bf16* bSrc = shdw  + (size_t)(h0 + lrow) * F_SH + c8;
  uint32_t aD[2] = { lds_off(&At[0][lrow * LDK + c8]), lds_off(&At[1][lrow * LDK + c8]) };
  uint32_t bD[2] = { lds_off(&Bt[0][lrow * LDK + c8]), lds_off(&Bt[1][lrow * LDK + c8]) };

  int mt = w & 3, nt = w >> 2;
  int aR  = 16 * mt + (lane & 15);
  int bR0 = 32 * nt + (lane & 15);
  int bR1 = bR0 + 16;

  v8f a0 = {0.f,0.f,0.f,0.f,0.f,0.f,0.f,0.f};
  v8f a1 = a0;

  async_cp16(aD[0], aSrc);
  async_cp16(bD[0], bSrc);
  async_wait_all();
  __syncthreads();

  int p = 0;
  for (int k0 = 0; k0 < F_SH; k0 += 32) {
    int q = p ^ 1;
    if (k0 + 32 < F_SH) {
      async_cp16(aD[q], aSrc + k0 + 32);
      async_cp16(bD[q], bSrc + k0 + 32);
    }
    v16bf a  = lds_fragA(At[p], aR, lane);
    v16bf b0 = lds_fragB(Bt[p], bR0, lane);
    v16bf b1 = lds_fragB(Bt[p], bR1, lane);
    a0 = wmma_bf16(a, b0, a0);
    a1 = wmma_bf16(a, b1, a1);
    async_wait_all();
    __syncthreads();
    p = q;
  }
  int half = lane >> 4, nl = lane & 15;
#pragma unroll
  for (int j = 0; j < 2; ++j) {
    v8f c = j ? a1 : a0;
#pragma unroll
    for (int r = 0; r < 8; ++r) {
      int t   = rowBase + 16 * mt + 8 * half + r;
      int col = h0 + 32 * nt + 16 * j + nl;
      float v = c[r];
#pragma unroll
      for (int k = 0; k < TOPK; ++k) {             // fixed order -> deterministic
        int slot = slot_of[t * TOPK + k];
        v += contrib[(size_t)slot * H_DIM + col];
      }
      out[(size_t)t * H_DIM + col] = v;
    }
  }
}

// ---------------- launch ----------------
extern "C" void kernel_launch(void* const* d_in, const int* in_sizes, int n_in,
                              void* d_out, int out_size, void* d_ws, size_t ws_size,
                              hipStream_t stream) {
  const float* hidden  = (const float*)d_in[0];
  const float* gate_wt = (const float*)d_in[1];
  const float* bias    = (const float*)d_in[2];
  const float* gate_w  = (const float*)d_in[3];
  const float* up_w    = (const float*)d_in[4];
  const float* down_w  = (const float*)d_in[5];
  const float* sh_gate = (const float*)d_in[6];
  const float* sh_up   = (const float*)d_in[7];
  const float* sh_down = (const float*)d_in[8];
  float* out = (float*)d_out;
  (void)in_sizes; (void)n_in; (void)out_size; (void)ws_size;

  char* base = (char*)d_ws;
  size_t off = 0;
  auto alloc = [&](size_t bytes) -> char* {
    char* r = base + off;
    off = (off + bytes + 255) & ~(size_t)255;
    return r;
  };
  int*    topk_idx   = (int*)   alloc((size_t)T_TOK * TOPK * sizeof(int));
  float*  topk_w     = (float*) alloc((size_t)T_TOK * TOPK * sizeof(float));
  int*    counts     = (int*)   alloc(NEXP * sizeof(int));
  int*    cursor     = (int*)   alloc(NEXP * sizeof(int));
  int*    offs       = (int*)   alloc((NEXP + 1) * sizeof(int));
  int*    slot_token = (int*)   alloc((size_t)T_TOK * TOPK * sizeof(int));
  float*  slot_w     = (float*) alloc((size_t)T_TOK * TOPK * sizeof(float));
  int*    slot_of    = (int*)   alloc((size_t)T_TOK * TOPK * sizeof(int));
  __bf16* act        = (__bf16*)alloc((size_t)T_TOK * TOPK * F_MOE_ * 2);
  __bf16* actsh      = (__bf16*)alloc((size_t)T_TOK * F_SH * 2);
  float*  contrib    = (float*) alloc((size_t)T_TOK * TOPK * H_DIM * sizeof(float));

  const size_t nHid  = (size_t)T_TOK * H_DIM;            // 8.4M
  const size_t nGate = (size_t)NEXP * F_MOE_ * H_DIM;    // 33.6M (gate/up/down)
  const size_t nSh   = (size_t)F_SH * H_DIM;             // 4.2M  (each shared)
  __bf16* hid_b = (__bf16*)alloc(nHid  * 2);
  __bf16* gw_b  = (__bf16*)alloc(nGate * 2);
  __bf16* uw_b  = (__bf16*)alloc(nGate * 2);
  __bf16* dw_b  = (__bf16*)alloc(nGate * 2);
  __bf16* shg_b = (__bf16*)alloc(nSh * 2);
  __bf16* shu_b = (__bf16*)alloc(nSh * 2);
  __bf16* shd_b = (__bf16*)alloc(nSh * 2);

  // one-time fp32 -> bf16 conversion (weights + hidden)
  cvt_f32_bf16_kernel<<<2048, 256, 0, stream>>>(hidden,  hid_b, (long)(nHid  / 4));
  cvt_f32_bf16_kernel<<<4096, 256, 0, stream>>>(gate_w,  gw_b,  (long)(nGate / 4));
  cvt_f32_bf16_kernel<<<4096, 256, 0, stream>>>(up_w,    uw_b,  (long)(nGate / 4));
  cvt_f32_bf16_kernel<<<4096, 256, 0, stream>>>(down_w,  dw_b,  (long)(nGate / 4));
  cvt_f32_bf16_kernel<<<1024, 256, 0, stream>>>(sh_gate, shg_b, (long)(nSh / 4));
  cvt_f32_bf16_kernel<<<1024, 256, 0, stream>>>(sh_up,   shu_b, (long)(nSh / 4));
  cvt_f32_bf16_kernel<<<1024, 256, 0, stream>>>(sh_down, shd_b, (long)(nSh / 4));

  // routing + dispatch
  moe_router_kernel<<<T_TOK, 256, 0, stream>>>(hidden, gate_wt, bias, topk_w, topk_idx);
  moe_zero_counts<<<1, 32, 0, stream>>>(counts);
  moe_count<<<(T_TOK * TOPK + 255) / 256, 256, 0, stream>>>(topk_idx, counts);
  moe_scan<<<1, 32, 0, stream>>>(counts, offs, cursor);
  moe_fill<<<(T_TOK * TOPK + 255) / 256, 256, 0, stream>>>(topk_idx, topk_w, cursor,
                                                           slot_token, slot_w, slot_of);
  // routed gate/up (gathered rows), expert-major for L2 weight reuse
  moe_gateup_kernel<<<dim3(F_MOE_ / 64, T_TOK / 64, NEXP), 256, 0, stream>>>(
      hid_b, gw_b, uw_b, slot_token, offs, act, F_MOE_, 1);
  // shared expert gate/up
  moe_gateup_kernel<<<dim3(F_SH / 64, T_TOK / 64, 1), 256, 0, stream>>>(
      hid_b, shg_b, shu_b, slot_token, offs, actsh, F_SH, 0);
  // routed down-proj -> weighted contributions
  moe_down_kernel<<<dim3(H_DIM / 64, T_TOK / 64, NEXP), 256, 0, stream>>>(
      act, dw_b, slot_w, offs, contrib);
  // shared down-proj + routed reduction -> output
  moe_final_kernel<<<dim3(H_DIM / 64, T_TOK / 64, 1), 256, 0, stream>>>(
      actsh, shd_b, contrib, slot_of, out);
}
